// SelfAttention_3891240370352
// MI455X (gfx1250) — compile-verified, matrix operand order checked by reading, not run on hardware
//
#include <hip/hip_runtime.h>
#include <hip/hip_bf16.h>

// Problem constants (match reference)
constexpr int Nn = 4;
constexpr int Ss = 2048;
constexpr int Ee = 1024;
constexpr int Hh = 16;
constexpr int Dd = 64;            // head dim
constexpr float SCALE = 0.03125f; // 1/sqrt(E) = 1/32

typedef __attribute__((ext_vector_type(16))) _Float16 v16h;
typedef __attribute__((ext_vector_type(8)))  _Float16 v8h;
typedef __attribute__((ext_vector_type(8)))  float    v8f;
typedef __attribute__((ext_vector_type(4)))  float    v4f;
typedef __attribute__((ext_vector_type(4)))  unsigned u32x4;
typedef __attribute__((ext_vector_type(4)))  int      i32x4;
typedef __attribute__((ext_vector_type(8)))  int      i32x8;

union V16HU { v16h v; v8h h[2]; };

#if __has_builtin(__builtin_amdgcn_tensor_load_to_lds)
#define USE_TDM 1
#else
#define USE_TDM 0
#endif

// A/B fragment load from row-major f16 (B^T row-major == same packing as A).
// Lane L: r = L&15, half = L>>4. elems[0..7]  = p[r*ld + kbase + half*8 + i]
//                               elems[8..15] = p[r*ld + kbase + 16 + half*8 + i]
__device__ inline v16h load_frag_f16(const _Float16* p, int ld, int kbase,
                                     int lane) {
  const int r = lane & 15, hf = (lane >> 4) & 1;
  const _Float16* base = p + (size_t)r * ld + kbase + hf * 8;
  V16HU u;
  u.h[0] = *(const v8h*)(base);
  u.h[1] = *(const v8h*)(base + 16);
  return u.v;
}

// Same fragment from an f32 source (convert on the fly)
__device__ inline v16h load_frag_f32(const float* __restrict__ p, int ld,
                                     int kbase, int lane) {
  const int r = lane & 15, hf = (lane >> 4) & 1;
  const float* base = p + (size_t)r * ld + kbase + hf * 8;
  v16h o;
#pragma unroll
  for (int i = 0; i < 8; ++i) {
    o[i]     = (_Float16)base[i];
    o[i + 8] = (_Float16)base[16 + i];
  }
  return o;
}

__device__ inline v8f wmma16(v16h a, v16h b, v8f c) {
  return __builtin_amdgcn_wmma_f32_16x16x32_f16(false, a, false, b,
                                                (short)0, c, false, false);
}

__device__ inline float rowmax16(float x) {
#pragma unroll
  for (int m = 1; m < 16; m <<= 1) x = fmaxf(x, __shfl_xor(x, m, 32));
  return x;
}
__device__ inline float rowsum16(float x) {
#pragma unroll
  for (int m = 1; m < 16; m <<= 1) x += __shfl_xor(x, m, 32);
  return x;
}

#if USE_TDM
// Tensor Data Mover: 2-D tile (row-major, 2-byte elems) -> LDS.
// D# built per CDNA5 ISA ch.8: group0 = {count/flags, lds_addr, global_addr,
// type=2}; group1 = {data_size, tensor dims, tile dims, dim0 stride}.
__device__ inline void tdm_load_2d(unsigned lds_off, const _Float16* gptr,
                                   unsigned tile0, unsigned tile1,
                                   unsigned tens0, unsigned tens1,
                                   unsigned long long stride0) {
  unsigned long long ga = (unsigned long long)(size_t)gptr;
  u32x4 g0 = { 1u,                                   // count=1, user D#
               lds_off,                              // lds_addr
               (unsigned)(ga & 0xffffffffull),
               (unsigned)((ga >> 32) & 0x01ffffffull) | (2u << 30) }; // type=2
  i32x8 g1 = { (int)(1u << 16),                       // data_size=1 (2B), mask=0
               (int)((tens0 & 0xffffu) << 16),        // tensor_dim0[15:0]
               (int)(((tens0 >> 16) & 0xffffu) | ((tens1 & 0xffffu) << 16)),
               (int)(((tens1 >> 16) & 0xffffu) | ((tile0 & 0xffffu) << 16)),
               (int)(tile1 & 0xffffu),                // tile_dim1, tile_dim2=0
               (int)(unsigned)(stride0 & 0xffffffffull),
               (int)(unsigned)((stride0 >> 32) & 0xffffull),
               0 };
  i32x4 gz = {0, 0, 0, 0};
#if defined(__clang_major__) && __clang_major__ >= 23
  i32x8 gz8 = {0, 0, 0, 0, 0, 0, 0, 0};
  __builtin_amdgcn_tensor_load_to_lds(g0, g1, gz, gz, gz8, 0);
#else
  __builtin_amdgcn_tensor_load_to_lds(g0, g1, gz, gz, 0);
#endif
}
#endif

// ---------------- kernel 0: Wo f32 -> f16 -------------------------------
__global__ __launch_bounds__(256) void cvt_wo_kernel(
    const float* __restrict__ Wo, _Float16* __restrict__ WoH) {
  int i = blockIdx.x * 256 + threadIdx.x;
  WoH[i] = (_Float16)Wo[i];
}

// ---------------- kernel 1: fused QKV head projections ------------------
// Qp,Kp: [N,H,S,D] f16 ; VpT: [N,H,D,S] f16. Stores staged through LDS so
// global writes are coalesced b128 bursts (Q/K block is one 8KB linear run).
__global__ __launch_bounds__(128) void proj_kernel(
    const float* __restrict__ values, const float* __restrict__ keys,
    const float* __restrict__ query,  const float* __restrict__ Wv,
    const float* __restrict__ Wk,     const float* __restrict__ Wq,
    _Float16* __restrict__ Qp, _Float16* __restrict__ Kp,
    _Float16* __restrict__ VpT) {
  __shared__ __align__(16) _Float16 stg[64 * 64];  // 8KB staging
  const int tid  = threadIdx.x;
  const int lane = tid & 31;
  const int wave = tid >> 5;  // 4 waves, 16 s-rows each
  const int sBlocks = Ss / 64;
  int nh = blockIdx.x / sBlocks;
  int sb = blockIdx.x % sBlocks;
  int n = nh / Hh, h = nh % Hh;
  int row0 = sb * 64 + wave * 16;
  const int col = lane & 15, hf = (lane >> 4) & 1;

#pragma unroll
  for (int t = 0; t < 3; ++t) {
    const float* X = (t == 0) ? query : (t == 1) ? keys : values;
    const float* W = (t == 0) ? Wq    : (t == 1) ? Wk   : Wv;
    const float* Xtile = X + ((size_t)n * Ss + row0) * Ee + h * Dd;
    v16h A0 = load_frag_f32(Xtile, Ee, 0,  lane);
    v16h A1 = load_frag_f32(Xtile, Ee, 32, lane);
    v8f acc[4];
#pragma unroll
    for (int e = 0; e < 4; ++e) {
      v16h B0 = load_frag_f32(W + e * 16 * Dd, Dd, 0,  lane);
      v16h B1 = load_frag_f32(W + e * 16 * Dd, Dd, 32, lane);
      v8f a = {};
      a = wmma16(A0, B0, a);
      a = wmma16(A1, B1, a);
      acc[e] = a;
    }
    if (t < 2) {
      // stage [s_local][d] then one contiguous 8KB store
#pragma unroll
      for (int e = 0; e < 4; ++e)
#pragma unroll
        for (int rr = 0; rr < 8; ++rr)
          stg[(wave * 16 + rr + 8 * hf) * 64 + e * 16 + col] =
              (_Float16)acc[e][rr];
      __syncthreads();
      _Float16* dst =
          ((t == 0) ? Qp : Kp) + (((size_t)(n * Hh + h)) * Ss + sb * 64) * Dd;
#pragma unroll
      for (int k = 0; k < 4; ++k) {
        int cid = tid + 128 * k;          // 512 x 16B chunks
        int row = cid >> 3, c8 = cid & 7;
        *(v8h*)(dst + row * 64 + c8 * 8) = *(const v8h*)(stg + row * 64 + c8 * 8);
      }
      __syncthreads();
    } else {
      // V transposed: stage [d][s_local]; per-lane writes are packed b128
#pragma unroll
      for (int e = 0; e < 4; ++e) {
        v8h tmp;
#pragma unroll
        for (int rr = 0; rr < 8; ++rr) tmp[rr] = (_Float16)acc[e][rr];
        *(v8h*)(stg + (e * 16 + col) * 64 + wave * 16 + 8 * hf) = tmp;
      }
      __syncthreads();
      _Float16* dstv = VpT + ((size_t)(n * Hh + h)) * Dd * (size_t)Ss + sb * 64;
#pragma unroll
      for (int k = 0; k < 4; ++k) {
        int cid = tid + 128 * k;
        int row = cid >> 3, c8 = cid & 7;  // row = d
        *(v8h*)(dstv + (size_t)row * Ss + c8 * 8) =
            *(const v8h*)(stg + row * 64 + c8 * 8);
      }
    }
  }
}

// ---------------- kernel 2: flash attention -----------------------------
// 8 waves x 16 q rows; K/V blocks staged to LDS by the Tensor Data Mover
// (double buffered, overlapped with compute), P transposed through LDS.
__global__ __launch_bounds__(256) void attn_kernel(
    const _Float16* __restrict__ Qp, const _Float16* __restrict__ Kp,
    const _Float16* __restrict__ VpT, const int* __restrict__ mask,
    _Float16* __restrict__ Of16) {
  __shared__ __align__(16) _Float16 Kbuf[2][64 * 64];  // 16KB
  __shared__ __align__(16) _Float16 Vbuf[2][64 * 64];  // 16KB
  __shared__ __align__(16) _Float16 Pst[8][16 * 64];   // 16KB per-wave staging
  const int tid  = threadIdx.x;
  const int lane = tid & 31;
  const int wave = tid >> 5;
  const int qBlocks = Ss / 128;
  int nh = blockIdx.x / qBlocks;
  int qb = blockIdx.x % qBlocks;
  int n = nh / Hh, h = nh % Hh;
  int qrow0 = qb * 128 + wave * 16;

  const _Float16* Qbase = Qp + (((size_t)(n * Hh + h)) * Ss + qrow0) * Dd;
  const _Float16* Kbase = Kp + ((size_t)(n * Hh + h)) * Ss * Dd;
  const _Float16* Vbase = VpT + ((size_t)(n * Hh + h)) * Dd * (size_t)Ss;
  const int* maskBase = mask + (size_t)n * Ss * Ss;

  v16h QA0 = load_frag_f16(Qbase, Dd, 0,  lane);
  v16h QA1 = load_frag_f16(Qbase, Dd, 32, lane);

  v8f O[4] = {};
  float mrun[8], lrun[8];
#pragma unroll
  for (int rr = 0; rr < 8; ++rr) { mrun[rr] = -3.0e38f; lrun[rr] = 0.0f; }

  const int col = lane & 15, hf = (lane >> 4) & 1;
  _Float16* myP = &Pst[wave][0];
  const int NB = Ss / 64;

#if USE_TDM
  if (wave == 0) {  // prologue DMA for block 0
    tdm_load_2d((unsigned)(size_t)&Kbuf[0][0], Kbase, 4096u, 1u, 4096u, 1u, 4096ull);
    tdm_load_2d((unsigned)(size_t)&Vbuf[0][0], Vbase, 64u, 64u,
                (unsigned)Ss, 64u, (unsigned long long)Ss);
  }
#endif

  for (int kb = 0; kb < NB; ++kb) {
    const int kcol0 = kb * 64;
    const int cur = kb & 1;
#if USE_TDM
    if (wave == 0) __builtin_amdgcn_s_wait_tensorcnt(0);
    __syncthreads();  // buf[cur] ready; everyone past reads of buf[cur^1]
    if (wave == 0 && kb + 1 < NB) {  // overlap next block's DMA with compute
      tdm_load_2d((unsigned)(size_t)&Kbuf[cur ^ 1][0],
                  Kbase + (size_t)(kcol0 + 64) * Dd, 4096u, 1u, 4096u, 1u,
                  4096ull);
      tdm_load_2d((unsigned)(size_t)&Vbuf[cur ^ 1][0], Vbase + kcol0 + 64,
                  64u, 64u, (unsigned)Ss, 64u, (unsigned long long)Ss);
    }
#else
    __syncthreads();
    // cooperative fallback staging
#pragma unroll
    for (int k = 0; k < 2; ++k) {
      int cid = tid + 256 * k;
      int row = cid >> 3, c8 = cid & 7;
      *(v8h*)(&Kbuf[cur][0] + row * 64 + c8 * 8) =
          *(const v8h*)(Kbase + (size_t)(kcol0 + row) * Dd + c8 * 8);
      *(v8h*)(&Vbuf[cur][0] + row * 64 + c8 * 8) =
          *(const v8h*)(Vbase + (size_t)row * Ss + kcol0 + c8 * 8);
    }
    __syncthreads();
#endif
    const _Float16* Kc = &Kbuf[cur][0];
    const _Float16* Vc = &Vbuf[cur][0];
    if (kb + 1 < NB)  // pull next mask rows toward the caches
      __builtin_prefetch(maskBase + (size_t)qrow0 * Ss + kcol0 + 64, 0, 1);

    // ---- scores: S = Q (16x64) @ K_blk^T (64x64), K from LDS ----
    v8f Sc[4];
#pragma unroll
    for (int j = 0; j < 4; ++j) {
      v16h KB0 = load_frag_f16(Kc + (j * 16) * 64, 64, 0,  lane);
      v16h KB1 = load_frag_f16(Kc + (j * 16) * 64, 64, 32, lane);
      v8f a = {};
      a = wmma16(QA0, KB0, a);
      a = wmma16(QA1, KB1, a);
      Sc[j] = a;
    }
    // ---- mask (before scale, like reference) + scale ----
#pragma unroll
    for (int j = 0; j < 4; ++j) {
      int kc = kcol0 + j * 16 + col;
#pragma unroll
      for (int rr = 0; rr < 8; ++rr) {
        int q = qrow0 + rr + 8 * hf;
        float e = Sc[j][rr];
        e = (maskBase[(size_t)q * Ss + kc] == 0) ? -1.0e10f : e;
        Sc[j][rr] = e * SCALE;
      }
    }
    // ---- online softmax ----
    float mnew[8], corr[8];
#pragma unroll
    for (int rr = 0; rr < 8; ++rr) {
      float t = fmaxf(fmaxf(Sc[0][rr], Sc[1][rr]), fmaxf(Sc[2][rr], Sc[3][rr]));
      t = rowmax16(t);
      mnew[rr] = fmaxf(mrun[rr], t);
      corr[rr] = __expf(mrun[rr] - mnew[rr]);
      mrun[rr] = mnew[rr];
    }
#pragma unroll
    for (int j = 0; j < 4; ++j) {
#pragma unroll
      for (int rr = 0; rr < 8; ++rr) {
        float p = __expf(Sc[j][rr] - mnew[rr]);
        Sc[j][rr] = p;
        myP[(rr + 8 * hf) * 64 + j * 16 + col] = (_Float16)p;  // C -> LDS
      }
    }
#pragma unroll
    for (int rr = 0; rr < 8; ++rr) {
      float t = Sc[0][rr] + Sc[1][rr] + Sc[2][rr] + Sc[3][rr];
      lrun[rr] = lrun[rr] * corr[rr] + rowsum16(t);
    }
#pragma unroll
    for (int d2 = 0; d2 < 4; ++d2)
#pragma unroll
      for (int rr = 0; rr < 8; ++rr) O[d2][rr] *= corr[rr];

    // wave-private P transpose bounce (same-wave LDS is HW in-order)
    __builtin_amdgcn_wave_barrier();
    v16h PA0 = load_frag_f16(myP, 64, 0,  lane);
    v16h PA1 = load_frag_f16(myP, 64, 32, lane);
    __builtin_amdgcn_wave_barrier();

    // ---- O += P (16x64) @ V_blk (64x64), V^T rows from LDS ----
#pragma unroll
    for (int d2 = 0; d2 < 4; ++d2) {
      const _Float16* vb = Vc + (d2 * 16) * 64;
      v16h VB0 = load_frag_f16(vb, 64, 0,  lane);
      v16h VB1 = load_frag_f16(vb, 64, 32, lane);
      O[d2] = wmma16(PA0, VB0, O[d2]);
      O[d2] = wmma16(PA1, VB1, O[d2]);
    }
  }

  // epilogue: normalize, stage 16x64 tile in LDS, coalesced b128 stores
#pragma unroll
  for (int rr = 0; rr < 8; ++rr) lrun[rr] = 1.0f / lrun[rr];
#pragma unroll
  for (int d2 = 0; d2 < 4; ++d2)
#pragma unroll
    for (int rr = 0; rr < 8; ++rr)
      myP[(rr + 8 * hf) * 64 + d2 * 16 + col] =
          (_Float16)(O[d2][rr] * lrun[rr]);
  __builtin_amdgcn_wave_barrier();
  size_t gbase = ((size_t)n * Ss + qrow0) * Ee + h * Dd;
#pragma unroll
  for (int k = 0; k < 4; ++k) {
    int cid = lane + 32 * k;           // 128 x 16B chunks
    int row = cid >> 3, c8 = cid & 7;
    *(v8h*)(Of16 + gbase + (size_t)row * Ee + c8 * 8) =
        *(const v8h*)(myP + row * 64 + c8 * 8);
  }
}

// ---------------- kernel 3: Y = X @ Wo^T + bo (f32 out) -----------------
__global__ __launch_bounds__(256) void outproj_kernel(
    const _Float16* __restrict__ X, const _Float16* __restrict__ WoH,
    const float* __restrict__ bo, float* __restrict__ Y) {
  __shared__ __align__(16) float ost[8][16 * 64];  // 32KB staging
  const int lane = threadIdx.x & 31;
  const int wave = threadIdx.x >> 5;
  const int colBlocks = Ee / 64;
  int rb = blockIdx.x / colBlocks;
  int cb = blockIdx.x % colBlocks;
  int row0 = rb * 128 + wave * 16;
  int col0 = cb * 64;
  const _Float16* Xbase = X + (size_t)row0 * Ee;

  v8f acc[4] = {};
  for (int kc = 0; kc < Ee / 32; ++kc) {
    v16h A = load_frag_f16(Xbase, Ee, kc * 32, lane);
#pragma unroll
    for (int j = 0; j < 4; ++j) {
      v16h B = load_frag_f16(WoH + (size_t)(col0 + j * 16) * Ee, Ee, kc * 32,
                             lane);
      acc[j] = wmma16(A, B, acc[j]);
    }
  }
  const int col = lane & 15, hf = (lane >> 4) & 1;
  float* sb = &ost[wave][0];
#pragma unroll
  for (int j = 0; j < 4; ++j) {
    float b = bo[col0 + j * 16 + col];
#pragma unroll
    for (int rr = 0; rr < 8; ++rr)
      sb[(rr + 8 * hf) * 64 + j * 16 + col] = acc[j][rr] + b;
  }
  __builtin_amdgcn_wave_barrier();
  size_t gbase = (size_t)row0 * Ee + col0;
#pragma unroll
  for (int k = 0; k < 8; ++k) {
    int cid = lane + 32 * k;            // 256 x 16B chunks
    int row = cid >> 4, c4 = cid & 15;
    *(v4f*)(Y + gbase + (size_t)row * Ee + c4 * 4) =
        *(const v4f*)(sb + row * 64 + c4 * 4);
  }
}

extern "C" void kernel_launch(void* const* d_in, const int* in_sizes, int n_in,
                              void* d_out, int out_size, void* d_ws,
                              size_t ws_size, hipStream_t stream) {
  (void)in_sizes; (void)n_in; (void)out_size; (void)ws_size;
  const float* values = (const float*)d_in[0];
  const float* keys   = (const float*)d_in[1];
  const float* query  = (const float*)d_in[2];
  const float* Wv     = (const float*)d_in[3];
  const float* Wk     = (const float*)d_in[4];
  const float* Wq     = (const float*)d_in[5];
  const float* Wo     = (const float*)d_in[6];
  const float* bo     = (const float*)d_in[7];
  const int*   mask   = (const int*)d_in[8];
  float* out = (float*)d_out;

  const size_t SZ_NHSD = (size_t)Nn * Hh * Ss * Dd * sizeof(_Float16);  // 16MB
  char* ws = (char*)d_ws;
  _Float16* Qp   = (_Float16*)(ws);
  _Float16* Kp   = (_Float16*)(ws + SZ_NHSD);
  _Float16* VpT  = (_Float16*)(ws + 2 * SZ_NHSD);
  _Float16* Of16 = (_Float16*)(ws + 3 * SZ_NHSD);
  _Float16* WoH  = (_Float16*)(ws + 4 * SZ_NHSD);

  cvt_wo_kernel<<<(Ee * Ee) / 256, 256, 0, stream>>>(Wo, WoH);
  proj_kernel<<<Nn * Hh * (Ss / 64), 128, 0, stream>>>(values, keys, query, Wv,
                                                       Wk, Wq, Qp, Kp, VpT);
  attn_kernel<<<Nn * Hh * (Ss / 128), 256, 0, stream>>>(Qp, Kp, VpT, mask,
                                                        Of16);
  outproj_kernel<<<(Nn * Ss / 128) * (Ee / 64), 256, 0, stream>>>(Of16, WoH, bo,
                                                                  out);
}
